// MolGNN_20753281974817
// MI455X (gfx1250) — matrix operand order repaired and checked by reading, not compiled, not run on hardware
//
#include <hip/hip_runtime.h>
#include <hip/hip_bf16.h>

// ---------------------------------------------------------------------------
// MolGNN on MI455X (gfx1250): WMMA bf16 GEMMs with fragment-ready packed
// weights, fused GRU gates on accumulator fragments, L2-resident edge scatter.
// ---------------------------------------------------------------------------

#define NN 200000   // nodes per component
#define EE 400000   // edges per component
#define HH 96       // hidden dim
#define LL 6        // GRU layers
#define BB 4096     // graphs per batch
#define FCD 384     // 4*H
#define FC1 1536    // 4*FC

typedef __attribute__((ext_vector_type(16))) __bf16 v16bf;
typedef __attribute__((ext_vector_type(8)))  float  v8f;

// ---- float -> bf16 (native convert, RNE) ----------------------------------
__device__ __forceinline__ __bf16 f2bf(float f) {
    return (__bf16)f;
}

__device__ __forceinline__ float sigmoidf_(float x) {
    return 1.0f / (1.0f + __expf(-x));
}

// ---- WMMA wrapper ----------------------------------------------------------
__device__ __forceinline__ v8f wmma_bf16(v16bf a, v16bf b, v8f c) {
    return __builtin_amdgcn_wmma_f32_16x16x32_bf16(
        /*neg_a=*/false, a, /*neg_b=*/false, b,
        /*c_mod=*/(short)0, c, /*reuse_a=*/false, /*reuse_b=*/false);
}

// ---- A fragment: 16x32 bf16 tile from f32 row-major source -----------------
// Per CDNA5 ISA layout: lane<16: M=lane, e=0..7 -> K(kb..kb+7), e=8..15 ->
// K(16+kb..16+kb+7) with kb=0; lane>=16: same with kb=8.
// Two contiguous 8-float runs per lane -> 4x float4 vector loads.
__device__ __forceinline__ v16bf load_a_f32(const float* __restrict__ A, int lda,
                                            int row0, int k0, int lane) {
    int m  = lane & 15;
    int kb = (lane >> 4) << 3;           // 0 or 8
    const float* r = A + (size_t)(row0 + m) * lda + k0;
    float4 x0 = *(const float4*)(r + kb);
    float4 x1 = *(const float4*)(r + kb + 4);
    float4 y0 = *(const float4*)(r + 16 + kb);
    float4 y1 = *(const float4*)(r + 16 + kb + 4);
    v16bf a;
    a[0]  = f2bf(x0.x); a[1]  = f2bf(x0.y); a[2]  = f2bf(x0.z); a[3]  = f2bf(x0.w);
    a[4]  = f2bf(x1.x); a[5]  = f2bf(x1.y); a[6]  = f2bf(x1.z); a[7]  = f2bf(x1.w);
    a[8]  = f2bf(y0.x); a[9]  = f2bf(y0.y); a[10] = f2bf(y0.z); a[11] = f2bf(y0.w);
    a[12] = f2bf(y1.x); a[13] = f2bf(y1.y); a[14] = f2bf(y1.z); a[15] = f2bf(y1.w);
    return a;
}

// ---- B fragment: packed fragment-ready layout ------------------------------
// Bpk[((nt*ktiles + kt)*32 + lane)*16 + e] : one contiguous 32-byte run per
// lane per fragment -> two global_load_b128.
__device__ __forceinline__ v16bf load_b_pk(const __bf16* __restrict__ Bpk,
                                           int ktiles, int kt, int nt, int lane) {
    const __bf16* p = Bpk + (((size_t)nt * ktiles + kt) * 32 + lane) * 16;
    return *(const v16bf*)p;
}

// C/D: 16x16 f32. lane<16: N=lane, VGPR v -> M=v. lane>=16: N=lane-16, M=8+v.
__device__ __forceinline__ void store_c(float* __restrict__ C, int ldc,
                                        int row0, int n0, int lane, v8f c) {
    int n  = lane & 15;
    int mb = (lane >> 4) << 3;
    float* p = C + (size_t)(row0 + mb) * ldc + (n0 + n);
#pragma unroll
    for (int v = 0; v < 8; ++v) p[(size_t)v * ldc] = c[v];
}

// K=96 accumulate helper over packed B (3 WMMA steps, ktiles==3)
__device__ __forceinline__ v8f gemm_k96(v16bf a0, v16bf a1, v16bf a2,
                                        const __bf16* __restrict__ Bpk,
                                        int nt, int lane) {
    v8f acc = {};
    acc = wmma_bf16(a0, load_b_pk(Bpk, 3, 0, nt, lane), acc);
    acc = wmma_bf16(a1, load_b_pk(Bpk, 3, 1, nt, lane), acc);
    acc = wmma_bf16(a2, load_b_pk(Bpk, 3, 2, nt, lane), acc);
    return acc;
}

// ---------------------------------------------------------------------------
// Weight packing: f32 [K,N] (or [N,K] with transpose) -> fragment-ready bf16
// ---------------------------------------------------------------------------
__global__ void molgnn_pack_b(const float* __restrict__ src, __bf16* __restrict__ dst,
                              int K, int N, int transpose) {
    int i = blockIdx.x * blockDim.x + threadIdx.x;
    if (i >= K * N) return;
    int e    = i & 15;
    int lane = (i >> 4) & 31;
    int tile = i >> 9;
    int ktiles = K >> 5;
    int kt = tile % ktiles;
    int nt = tile / ktiles;
    int n = nt * 16 + (lane & 15);
    int k = kt * 32 + ((lane >> 4) << 3) + (e & 7) + ((e >> 3) << 4);
    float v = transpose ? src[(size_t)n * K + k] : src[(size_t)k * N + n];
    dst[i] = f2bf(v);
}

// ---------------------------------------------------------------------------
// Kernels
// ---------------------------------------------------------------------------

// m = h @ W  (N x 96 . 96 x 96). 4 waves/block, 16 rows per wave.
__global__ void molgnn_msg(const float* __restrict__ h,
                           const __bf16* __restrict__ Wpk,
                           float* __restrict__ m) {
    int lane = threadIdx.x & 31;
    int row0 = (blockIdx.x * 4 + (threadIdx.x >> 5)) * 16;
    v16bf a0 = load_a_f32(h, HH, row0, 0,  lane);
    v16bf a1 = load_a_f32(h, HH, row0, 32, lane);
    v16bf a2 = load_a_f32(h, HH, row0, 64, lane);
#pragma unroll
    for (int nt = 0; nt < 6; ++nt) {
        v8f acc = gemm_k96(a0, a1, a2, Wpk, nt, lane);
        store_c(m, HH, row0, nt * 16, lane, acc);
    }
}

// agg[dst] += m[src] : one wave per edge, 3 features per lane (L2-resident)
__global__ void molgnn_scatter(const float* __restrict__ m,
                               const int* __restrict__ ei,
                               float* __restrict__ agg) {
    int wave = (blockIdx.x * blockDim.x + threadIdx.x) >> 5;
    int lane = threadIdx.x & 31;
    if (wave >= EE) return;
    int s = ei[wave];
    int d = ei[EE + wave];
    const float* mr = m + (size_t)s * HH;
    float* ar = agg + (size_t)d * HH;
    atomicAdd(&ar[lane],      mr[lane]);
    atomicAdd(&ar[lane + 32], mr[lane + 32]);
    atomicAdd(&ar[lane + 64], mr[lane + 64]);
}

// Fused GRU cell: gi = agg@wihT, gh = h@whhT via WMMA fragments; gates on
// accumulator registers; h updated in place (rows disjoint per wave).
// wihPk/whhPk: packed logical [96, 288] (ktiles=3, ntiles=18).
__global__ void molgnn_gru(float* __restrict__ h,
                           const float* __restrict__ agg,
                           const __bf16* __restrict__ wihPk,
                           const __bf16* __restrict__ whhPk,
                           const float* __restrict__ bih,
                           const float* __restrict__ bhh) {
    int lane = threadIdx.x & 31;
    int row0 = (blockIdx.x * 4 + (threadIdx.x >> 5)) * 16;

    v16bf g0 = load_a_f32(agg, HH, row0, 0,  lane);
    v16bf g1 = load_a_f32(agg, HH, row0, 32, lane);
    v16bf g2 = load_a_f32(agg, HH, row0, 64, lane);
    v16bf h0 = load_a_f32(h,   HH, row0, 0,  lane);
    v16bf h1 = load_a_f32(h,   HH, row0, 32, lane);
    v16bf h2 = load_a_f32(h,   HH, row0, 64, lane);

    int n  = lane & 15;
    int mb = (lane >> 4) << 3;

#pragma unroll
    for (int j = 0; j < 6; ++j) {
        int f = j * 16 + n;                          // output feature column
        v8f gir = gemm_k96(g0, g1, g2, wihPk, j,      lane);
        v8f giz = gemm_k96(g0, g1, g2, wihPk, j + 6,  lane);
        v8f gin = gemm_k96(g0, g1, g2, wihPk, j + 12, lane);
        v8f ghr = gemm_k96(h0, h1, h2, whhPk, j,      lane);
        v8f ghz = gemm_k96(h0, h1, h2, whhPk, j + 6,  lane);
        v8f ghn = gemm_k96(h0, h1, h2, whhPk, j + 12, lane);

        float bri = bih[f], bzi = bih[HH + f], bni = bih[2 * HH + f];
        float brh = bhh[f], bzh = bhh[HH + f], bnh = bhh[2 * HH + f];
        float* hp = h + (size_t)(row0 + mb) * HH + f;
#pragma unroll
        for (int v = 0; v < 8; ++v) {
            float r  = sigmoidf_(gir[v] + bri + ghr[v] + brh);
            float z  = sigmoidf_(giz[v] + bzi + ghz[v] + bzh);
            float nn = tanhf(gin[v] + bni + r * (ghn[v] + bnh));
            float ho = hp[(size_t)v * HH];
            hp[(size_t)v * HH] = (1.0f - z) * nn + z * ho;
        }
    }
}

// Generic WMMA GEMM: C[M,N] = act(A[M,K](f32->bf16) @ Bpk + bias)
__global__ void molgnn_gemm(const float* __restrict__ A,
                            const __bf16* __restrict__ Bpk,
                            const float* __restrict__ bias,
                            float* __restrict__ C,
                            int N, int K, int relu) {
    int lane = threadIdx.x & 31;
    int row0 = (blockIdx.y * 4 + (threadIdx.x >> 5)) * 16;
    int nt   = blockIdx.x;
    int ktiles = K >> 5;
    v8f acc = {};
    for (int kt = 0; kt < ktiles; ++kt) {
        v16bf a = load_a_f32(A, K, row0, kt * 32, lane);
        v16bf b = load_b_pk(Bpk, ktiles, kt, nt, lane);
        acc = wmma_bf16(a, b, acc);
    }
    int nn = lane & 15;
    int mb = (lane >> 4) << 3;
    float bv = bias[nt * 16 + nn];
    float* p = C + (size_t)(row0 + mb) * N + (nt * 16 + nn);
#pragma unroll
    for (int v = 0; v < 8; ++v) {
        float o = acc[v] + bv;
        if (relu) o = fmaxf(o, 0.0f);
        p[(size_t)v * N] = o;
    }
}

// ---- small helper kernels --------------------------------------------------
__global__ void molgnn_init_h(const float* __restrict__ x, float* __restrict__ h) {
    int i = blockIdx.x * blockDim.x + threadIdx.x;
    if (i < NN * HH) {
        int node = i / HH, f = i % HH;
        h[i] = (f < 32) ? x[node * 32 + f] : 0.0f;
    }
}

// relu + per-graph sum (one wave per node, 3 features per lane)
__global__ void molgnn_pool(const float* __restrict__ h, const int* __restrict__ batch,
                            float* __restrict__ sums, float* __restrict__ cnt) {
    int wave = (blockIdx.x * blockDim.x + threadIdx.x) >> 5;
    int lane = threadIdx.x & 31;
    if (wave >= NN) return;
    int b = batch[wave];
    const float* hr = h + (size_t)wave * HH;
    float* sr = sums + (size_t)b * HH;
    atomicAdd(&sr[lane],      fmaxf(hr[lane],      0.0f));
    atomicAdd(&sr[lane + 32], fmaxf(hr[lane + 32], 0.0f));
    atomicAdd(&sr[lane + 64], fmaxf(hr[lane + 64], 0.0f));
    if (lane == 0) atomicAdd(&cnt[b], 1.0f);
}

__global__ void molgnn_pool_fin(const float* __restrict__ sums,
                                const float* __restrict__ cnt,
                                float* __restrict__ g) {
    int i = blockIdx.x * blockDim.x + threadIdx.x;
    if (i < BB * HH) g[i] = sums[i] / fmaxf(cnt[i / HH], 1.0f);
}

__global__ void molgnn_concat(const float* __restrict__ g1, const float* __restrict__ g2,
                              const float* __restrict__ g3, float* __restrict__ X) {
    int i = blockIdx.x * blockDim.x + threadIdx.x;
    if (i < BB * HH) {
        int b = i / HH, f = i % HH;
        float a = g1[i], c = g2[i], d = g3[i];
        float* xr = X + (size_t)b * FCD + f;
        xr[0] = a; xr[HH] = c; xr[2 * HH] = d; xr[3 * HH] = a * c * d;
    }
}

__global__ void molgnn_fc3(const float* __restrict__ Y, const float* __restrict__ w,
                           const float* __restrict__ b, float* __restrict__ out) {
    int i = blockIdx.x * blockDim.x + threadIdx.x;
    if (i < BB * 3) {
        int r = i / 3, c = i % 3;
        const float* yr = Y + (size_t)r * FCD;
        float s = b[c];
        for (int k = 0; k < FCD; ++k) s += yr[k] * w[k * 3 + c];
        out[i] = s;
    }
}

// ---------------------------------------------------------------------------
// Launch
// ---------------------------------------------------------------------------
extern "C" void kernel_launch(void* const* d_in, const int* in_sizes, int n_in,
                              void* d_out, int out_size, void* d_ws, size_t ws_size,
                              hipStream_t stream) {
    (void)in_sizes; (void)n_in; (void)out_size; (void)ws_size;

    // ---- workspace carve (~276 MB total) ----
    size_t off = 0;
    auto carve = [&](size_t bytes) -> void* {
        void* p = (char*)d_ws + off;
        off += (bytes + 255) & ~(size_t)255;
        return p;
    };
    float*  h     = (float*)carve((size_t)NN * HH * 4);
    float*  m     = (float*)carve((size_t)NN * HH * 4);
    float*  agg   = (float*)carve((size_t)NN * HH * 4);
    __bf16* Wpk   = (__bf16*)carve((size_t)LL * HH * HH * 2);
    __bf16* wihPk = (__bf16*)carve((size_t)HH * 3 * HH * 2);
    __bf16* whhPk = (__bf16*)carve((size_t)HH * 3 * HH * 2);
    float*  gpool = (float*)carve((size_t)3 * BB * HH * 4);
    float*  sums  = (float*)carve((size_t)BB * HH * 4);
    float*  cnt   = (float*)carve((size_t)BB * 4);
    float*  X     = (float*)carve((size_t)BB * FCD * 4);
    float*  Y1    = (float*)carve((size_t)BB * FC1 * 4);
    float*  Y2    = (float*)carve((size_t)BB * FCD * 4);
    __bf16* fc1pk = (__bf16*)carve((size_t)FCD * FC1 * 2);
    __bf16* fc2pk = (__bf16*)carve((size_t)FC1 * FCD * 2);

    // ---- FC weight packing (once per call) ----
    molgnn_pack_b<<<(FCD * FC1 + 255) / 256, 256, 0, stream>>>(
        (const float*)d_in[24], fc1pk, FCD, FC1, 0);
    molgnn_pack_b<<<(FC1 * FCD + 255) / 256, 256, 0, stream>>>(
        (const float*)d_in[26], fc2pk, FC1, FCD, 0);

    // ---- components (inputs in setup_inputs() dict order, 8 per component) ----
    for (int c = 0; c < 3; ++c) {
        const float* x     = (const float*)d_in[8 * c + 0];
        const int*   ei    = (const int*)  d_in[8 * c + 1];
        const int*   batch = (const int*)  d_in[8 * c + 2];
        const float* W     = (const float*)d_in[8 * c + 3];
        const float* wih   = (const float*)d_in[8 * c + 4];
        const float* whh   = (const float*)d_in[8 * c + 5];
        const float* bih   = (const float*)d_in[8 * c + 6];
        const float* bhh   = (const float*)d_in[8 * c + 7];

        for (int l = 0; l < LL; ++l)
            molgnn_pack_b<<<(HH * HH + 255) / 256, 256, 0, stream>>>(
                W + (size_t)l * HH * HH, Wpk + (size_t)l * HH * HH, HH, HH, 0);
        molgnn_pack_b<<<(HH * 3 * HH + 255) / 256, 256, 0, stream>>>(
            wih, wihPk, HH, 3 * HH, 1);
        molgnn_pack_b<<<(HH * 3 * HH + 255) / 256, 256, 0, stream>>>(
            whh, whhPk, HH, 3 * HH, 1);
        molgnn_init_h<<<(NN * HH + 255) / 256, 256, 0, stream>>>(x, h);

        for (int l = 0; l < LL; ++l) {
            molgnn_msg<<<NN / 64, 128, 0, stream>>>(h, Wpk + (size_t)l * HH * HH, m);
            hipMemsetAsync(agg, 0, (size_t)NN * HH * 4, stream);
            molgnn_scatter<<<EE / 8, 256, 0, stream>>>(m, ei, agg);
            molgnn_gru<<<NN / 64, 128, 0, stream>>>(h, agg, wihPk, whhPk, bih, bhh);
        }

        hipMemsetAsync(sums, 0, (size_t)BB * HH * 4, stream);
        hipMemsetAsync(cnt,  0, (size_t)BB * 4, stream);
        molgnn_pool<<<NN / 8, 256, 0, stream>>>(h, batch, sums, cnt);
        molgnn_pool_fin<<<(BB * HH + 255) / 256, 256, 0, stream>>>(sums, cnt,
                                                                   gpool + (size_t)c * BB * HH);
    }

    // ---- head ----
    molgnn_concat<<<(BB * HH + 255) / 256, 256, 0, stream>>>(
        gpool, gpool + (size_t)BB * HH, gpool + (size_t)2 * BB * HH, X);

    molgnn_gemm<<<dim3(FC1 / 16, BB / 64), 128, 0, stream>>>(
        X, fc1pk, (const float*)d_in[25], Y1, FC1, FCD, 1);
    molgnn_gemm<<<dim3(FCD / 16, BB / 64), 128, 0, stream>>>(
        Y1, fc2pk, (const float*)d_in[27], Y2, FCD, FC1, 1);
    molgnn_fc3<<<(BB * 3 + 255) / 256, 256, 0, stream>>>(
        Y2, (const float*)d_in[28], (const float*)d_in[29], (float*)d_out);
}
